// MultiHeadAttention_51127290691781
// MI455X (gfx1250) — matrix operand order repaired and checked by reading, not compiled, not run on hardware
//
#include <hip/hip_runtime.h>
#include <hip/hip_bf16.h>
#include <math.h>

// ---------------------------------------------------------------------------
// MHA forward for MI455X (gfx1250, wave32, WMMA).
// Bandwidth-bound on the [B,H,S,S] score tensor (268MB written + 268MB bias
// read + softmax pass); all GEMMs use v_wmma_f32_16x16x32_f16 (f32 acc).
// f16-source LDS staging uses GLOBAL_LOAD_ASYNC_TO_LDS_B128 (ASYNCcnt),
// f32->f16 staging uses global_load_b128 + v_cvt_pk_f16_f32 + ds_store.
// ---------------------------------------------------------------------------

#define HID    512
#define NHEADS 8
#define DHEAD  64
#define BATCH  2
#define SEQ    2048
#define MTOT   (BATCH * SEQ)                     // 4096 rows for projections

#define OUT0_ELEMS   ((size_t)BATCH * SEQ * HID)            // 2,097,152 f32
#define HALF_MAT     ((size_t)BATCH * NHEADS * SEQ * DHEAD) // 2,097,152 f16

typedef __attribute__((ext_vector_type(16))) _Float16 v16h;
typedef __attribute__((ext_vector_type(8)))  _Float16 v8h;
typedef __attribute__((ext_vector_type(8)))  float    v8f;

union V16U { v16h v; v8h h[2]; };
union AccU { v8f  v; float f[8]; };

// ---- CDNA5 async copy: global -> LDS, 16B per lane, tracked by ASYNCcnt ----
__device__ __forceinline__ void lds_async_b128(void* lds, const void* g) {
  unsigned l = (unsigned)(size_t)lds;     // low 32 bits of generic = LDS addr
  asm volatile("global_load_async_to_lds_b128 %0, %1, off"
               :: "v"(l), "v"(g) : "memory");
}
__device__ __forceinline__ void wait_async0() {
  asm volatile("s_wait_asynccnt 0" ::: "memory");
}

// ---- WMMA fragment gathers from LDS (ISA 7.12.2 layouts) -------------------
// A (16x32 f16, row-major tile, ld in halves): lane L -> row m = L%16,
// element e -> K = 16*(e/8) + 8*(L/16) + (e%8). Two 16B ds loads per lane.
__device__ __forceinline__ v16h frag_a(const _Float16* base, int ld, int lane) {
  const int m  = lane & 15;
  const int hs = lane >> 4;
  const _Float16* p = base + m * ld + hs * 8;
  V16U r;
  r.h[0] = *(const v8h*)(p);
  r.h[1] = *(const v8h*)(p + 16);
  return r.v;
}
// B (32x16 f16) gathered from an N-major tile Bt[n][k] (ld in halves):
// lane L -> col n = L%16, element e -> K = 16*(e/8) + 8*(L/16) + (e%8). Two 16B ds loads.
__device__ __forceinline__ v16h frag_b(const _Float16* base, int ld, int lane) {
  const int n  = lane & 15;
  const int hs = lane >> 4;
  const _Float16* p = base + n * ld + hs * 8;
  V16U r;
  r.h[0] = *(const v8h*)(p);
  r.h[1] = *(const v8h*)(p + 16);
  return r.v;
}

__device__ __forceinline__ v8f wmma_f16(v16h a, v16h b, v8f c) {
  return __builtin_amdgcn_wmma_f32_16x16x32_f16(false, a, false, b,
                                                (short)0, c, false, false);
}

// f32->f16 convert 8 values from two float4s
__device__ __forceinline__ v8h cvt8(float4 f0, float4 f1) {
  v8h h;
  h[0] = (_Float16)f0.x; h[1] = (_Float16)f0.y;
  h[2] = (_Float16)f0.z; h[3] = (_Float16)f0.w;
  h[4] = (_Float16)f1.x; h[5] = (_Float16)f1.y;
  h[6] = (_Float16)f1.z; h[7] = (_Float16)f1.w;
  return h;
}

// ---------------------------------------------------------------------------
// Kernel 1: Q/K/V projections.  Y = X @ W + bias, X:[4096,512] f32, W:[512,512].
// blockIdx.z selects the projection. Output layouts (f16, in workspace):
//   p=0: Qh[b][h][s][d]   p=1: Kh[b][h][s][d]   p=2: VT[b][h][d][s]
// Block = 256 thr (8 waves), tile 64(M) x 32(N), K-step 64 (2 WMMA / stage).
// ---------------------------------------------------------------------------
__global__ void __launch_bounds__(256)
k_qkv_proj(const float* __restrict__ q, const float* __restrict__ k,
           const float* __restrict__ v,
           const float* __restrict__ Wq, const float* __restrict__ bq,
           const float* __restrict__ Wk, const float* __restrict__ bk,
           const float* __restrict__ Wv, const float* __restrict__ bv,
           _Float16* __restrict__ Qh, _Float16* __restrict__ Kh,
           _Float16* __restrict__ VT) {
  __shared__ __align__(16) _Float16 At[64 * 72];   // 64 rows x 64 K (+8 pad)
  __shared__ __align__(16) _Float16 Bt[32 * 72];   // 32 N x 64 K (N-major)

  const int p     = blockIdx.z;
  const float* X  = (p == 0) ? q  : (p == 1) ? k  : v;
  const float* W  = (p == 0) ? Wq : (p == 1) ? Wk : Wv;
  const float* bs = (p == 0) ? bq : (p == 1) ? bk : bv;

  const int mBase = blockIdx.x * 64;
  const int nBase = blockIdx.y * 32;
  const int tid   = threadIdx.x;
  const int lane  = tid & 31;
  const int wave  = tid >> 5;
  const int mt    = wave >> 1;            // 0..3
  const int nt    = wave & 1;             // 0..1

  v8f acc = {};
  for (int kb = 0; kb < HID; kb += 64) {
    __syncthreads();
    { // stage A: 64x64 f32 -> f16, 16 elems/thread
      const int row = tid >> 2, c16 = (tid & 3) * 16;
      const float4* src =
          (const float4*)(X + (size_t)(mBase + row) * HID + kb + c16);
      float4 f0 = src[0], f1 = src[1], f2 = src[2], f3 = src[3];
      *(v8h*)&At[row * 72 + c16]     = cvt8(f0, f1);
      *(v8h*)&At[row * 72 + c16 + 8] = cvt8(f2, f3);
    }
    { // stage Bt[n][k] = W[kb+k][nBase+n] (transpose while staging), 8/thread
      const int n = tid & 31, k8 = (tid >> 5) * 8;
      #pragma unroll
      for (int j = 0; j < 8; ++j)
        Bt[n * 72 + k8 + j] =
            (_Float16)W[(size_t)(kb + k8 + j) * HID + nBase + n];
    }
    __syncthreads();
    acc = wmma_f16(frag_a(&At[(mt * 16) * 72], 72, lane),
                   frag_b(&Bt[(nt * 16) * 72], 72, lane), acc);
    acc = wmma_f16(frag_a(&At[(mt * 16) * 72 + 32], 72, lane),
                   frag_b(&Bt[(nt * 16) * 72 + 32], 72, lane), acc);
  }

  // epilogue: add bias, scatter to head-major f16 layouts
  AccU u; u.v = acc;
  const int hs = lane >> 4;
  const int n  = nBase + nt * 16 + (lane & 15);
  const float bias = bs[n];
  const int h = n >> 6, d = n & 63;
  #pragma unroll
  for (int i = 0; i < 8; ++i) {
    const int m  = mBase + mt * 16 + i + 8 * hs;
    const int bb = m >> 11, s = m & (SEQ - 1);
    const _Float16 val = (_Float16)(u.f[i] + bias);
    const int bh = bb * NHEADS + h;
    if (p == 0)      Qh[((size_t)bh * SEQ + s) * DHEAD + d] = val;
    else if (p == 1) Kh[((size_t)bh * SEQ + s) * DHEAD + d] = val;
    else             VT[((size_t)bh * DHEAD + d) * SEQ + s] = val;
  }
}

// ---------------------------------------------------------------------------
// Kernel 2: raw scores = Qh·Khᵀ * D^-0.5 + bias, masked -> f32 to d_out.
// Per (b,h): tile 64(q) x 64(k); each wave owns one 16-row strip and two
// 16-col accumulators (A-fragment reuse). Q/K tiles staged via async-to-LDS.
// ---------------------------------------------------------------------------
__global__ void __launch_bounds__(256)
k_scores(const _Float16* __restrict__ Qh, const _Float16* __restrict__ Kh,
         const float* __restrict__ attn_bias,
         const unsigned char* __restrict__ attn_mask,
         float* __restrict__ scores) {
  __shared__ __align__(16) _Float16 Qt[64 * 72];   // 64 q x 64 d (+8 pad)
  __shared__ __align__(16) _Float16 Kt[64 * 72];   // 64 k x 64 d

  const int qBase = blockIdx.x * 64;
  const int kBase = blockIdx.y * 64;
  const int bh    = blockIdx.z;                    // b*8 + h
  const int b     = bh >> 3;
  const int tid   = threadIdx.x;
  const int lane  = tid & 31;
  const int wave  = tid >> 5;
  const int mt    = wave >> 1, ntp = wave & 1;     // ntp: which 32-col pair

  // async stage: 64x64 f16 tiles = 512 x 16B chunks each; 2 chunks per thread
  #pragma unroll
  for (int c = tid; c < 512; c += 256) {
    const int row = c >> 3, c8 = (c & 7) * 8;
    lds_async_b128(&Qt[row * 72 + c8],
                   Qh + ((size_t)bh * SEQ + qBase + row) * DHEAD + c8);
    lds_async_b128(&Kt[row * 72 + c8],
                   Kh + ((size_t)bh * SEQ + kBase + row) * DHEAD + c8);
  }
  wait_async0();
  __syncthreads();

  v8f acc0 = {}, acc1 = {};
  #pragma unroll
  for (int kk = 0; kk < 64; kk += 32) {
    v16h a  = frag_a(&Qt[(mt * 16) * 72 + kk], 72, lane);
    v16h b0 = frag_b(&Kt[(ntp * 32) * 72 + kk], 72, lane);
    v16h b1 = frag_b(&Kt[(ntp * 32 + 16) * 72 + kk], 72, lane);
    acc0 = wmma_f16(a, b0, acc0);
    acc1 = wmma_f16(a, b1, acc1);
  }

  const int hs = lane >> 4;
  const float scale = 0.125f;                      // 64^-0.5
  AccU u[2]; u[0].v = acc0; u[1].v = acc1;
  #pragma unroll
  for (int jj = 0; jj < 2; ++jj) {
    const int kk = kBase + ntp * 32 + jj * 16 + (lane & 15);
    #pragma unroll
    for (int i = 0; i < 8; ++i) {
      const int qq = qBase + mt * 16 + i + 8 * hs;
      const size_t sidx = ((size_t)bh * SEQ + qq) * SEQ + kk;
      float val = u[jj].f[i] * scale + attn_bias[sidx];
      if (attn_mask[((size_t)b * SEQ + qq) * SEQ + kk]) val = -1e9f;
      scores[sidx] = val;
    }
  }
}

// ---------------------------------------------------------------------------
// Kernel 3: in-place softmax over the last axis (2048) of scores.
// One block (256 thr) per row; 8 strided elems per thread.
// ---------------------------------------------------------------------------
__global__ void __launch_bounds__(256)
k_softmax(float* __restrict__ scores) {
  __shared__ float red[256];
  const int t = threadIdx.x;
  float* row = scores + (size_t)blockIdx.x * SEQ;

  float vloc[8];
  float m = -INFINITY;
  #pragma unroll
  for (int j = 0; j < 8; ++j) {
    vloc[j] = row[t + j * 256];
    m = fmaxf(m, vloc[j]);
  }
  red[t] = m; __syncthreads();
  for (int s = 128; s > 0; s >>= 1) {
    if (t < s) red[t] = fmaxf(red[t], red[t + s]);
    __syncthreads();
  }
  const float rmax = red[0];
  __syncthreads();

  float sum = 0.f;
  #pragma unroll
  for (int j = 0; j < 8; ++j) {
    vloc[j] = __expf(vloc[j] - rmax);
    sum += vloc[j];
  }
  red[t] = sum; __syncthreads();
  for (int s = 128; s > 0; s >>= 1) {
    if (t < s) red[t] += red[t + s];
    __syncthreads();
  }
  const float inv = 1.0f / red[0];
  #pragma unroll
  for (int j = 0; j < 8; ++j) row[t + j * 256] = vloc[j] * inv;
}

// ---------------------------------------------------------------------------
// Kernel 4: ctx = P · V.  Per (b,h): M=q, N=d(64), K=keys(2048).
// P staged f32->f16 from d_out; VT tile (Bᵀ-contiguous f16) staged async.
// Writes ctx f16 [4096 x 512] row-major (input to out-projection).
// ---------------------------------------------------------------------------
__global__ void __launch_bounds__(256)
k_ctx(const float* __restrict__ scores, const _Float16* __restrict__ VT,
      _Float16* __restrict__ Ctx) {
  __shared__ __align__(16) _Float16 At[64 * 40];   // 64 q x 32 keys
  __shared__ __align__(16) _Float16 Bt[32 * 40];   // 32 d x 32 keys (N-major)

  const int qBase = blockIdx.x * 64;
  const int dBase = blockIdx.y * 32;
  const int bh    = blockIdx.z;
  const int b     = bh >> 3, h = bh & 7;
  const int tid   = threadIdx.x;
  const int lane  = tid & 31;
  const int wave  = tid >> 5;
  const int mt    = wave >> 1, nt = wave & 1;

  v8f acc = {};
  for (int kb = 0; kb < SEQ; kb += 32) {
    __syncthreads();
    { // stage P: 64x32 f32 -> f16
      const int row = tid >> 2, c8 = (tid & 3) * 8;
      const float4* src = (const float4*)(scores +
          ((size_t)bh * SEQ + qBase + row) * SEQ + kb + c8);
      float4 f0 = src[0], f1 = src[1];
      *(v8h*)&At[row * 40 + c8] = cvt8(f0, f1);
    }
    if (tid < 128) { // stage VT tile async: 32 rows x 32 halves = 128 chunks
      const int n = tid >> 2, c8 = (tid & 3) * 8;
      lds_async_b128(&Bt[n * 40 + c8],
                     VT + ((size_t)bh * DHEAD + dBase + n) * SEQ + kb + c8);
    }
    wait_async0();
    __syncthreads();
    acc = wmma_f16(frag_a(&At[(mt * 16) * 40], 40, lane),
                   frag_b(&Bt[(nt * 16) * 40], 40, lane), acc);
  }

  AccU u; u.v = acc;
  const int hs = lane >> 4;
  const int d  = dBase + nt * 16 + (lane & 15);
  #pragma unroll
  for (int i = 0; i < 8; ++i) {
    const int qq = qBase + mt * 16 + i + 8 * hs;
    Ctx[((size_t)(b * SEQ + qq)) * HID + h * DHEAD + d] = (_Float16)u.f[i];
  }
}

// ---------------------------------------------------------------------------
// Kernel 5: out = ctx @ Wo + bo  -> f32 region 0 of d_out.
// ctx tiles (f16) staged async; Wo staged with f32->f16 conversion.
// ---------------------------------------------------------------------------
__global__ void __launch_bounds__(256)
k_out_proj(const _Float16* __restrict__ Ctx, const float* __restrict__ Wo,
           const float* __restrict__ bo, float* __restrict__ out) {
  __shared__ __align__(16) _Float16 At[64 * 72];   // 64 m x 64 k
  __shared__ __align__(16) _Float16 Bt[32 * 72];   // 32 n x 64 k (N-major)

  const int mBase = blockIdx.x * 64;
  const int nBase = blockIdx.y * 32;
  const int tid   = threadIdx.x;
  const int lane  = tid & 31;
  const int wave  = tid >> 5;
  const int mt    = wave >> 1, nt = wave & 1;

  v8f acc = {};
  for (int kb = 0; kb < HID; kb += 64) {
    __syncthreads();
    { // stage ctx tile async: 64x64 halves = 512 chunks, 2 per thread
      #pragma unroll
      for (int c = tid; c < 512; c += 256) {
        const int row = c >> 3, c8 = (c & 7) * 8;
        lds_async_b128(&At[row * 72 + c8],
                       Ctx + (size_t)(mBase + row) * HID + kb + c8);
      }
    }
    { // stage Woᵀ tile: f32 -> f16, 8 strided reads per thread
      const int n = tid & 31, k8 = (tid >> 5) * 8;
      #pragma unroll
      for (int j = 0; j < 8; ++j)
        Bt[n * 72 + k8 + j] =
            (_Float16)Wo[(size_t)(kb + k8 + j) * HID + nBase + n];
    }
    wait_async0();
    __syncthreads();
    acc = wmma_f16(frag_a(&At[(mt * 16) * 72], 72, lane),
                   frag_b(&Bt[(nt * 16) * 72], 72, lane), acc);
    acc = wmma_f16(frag_a(&At[(mt * 16) * 72 + 32], 72, lane),
                   frag_b(&Bt[(nt * 16) * 72 + 32], 72, lane), acc);
  }

  AccU u; u.v = acc;
  const int hs = lane >> 4;
  const int n  = nBase + nt * 16 + (lane & 15);
  const float bias = bo[n];
  #pragma unroll
  for (int i = 0; i < 8; ++i) {
    const int m = mBase + mt * 16 + i + 8 * hs;
    out[(size_t)m * HID + n] = u.f[i] + bias;
  }
}

// ---------------------------------------------------------------------------
extern "C" void kernel_launch(void* const* d_in, const int* in_sizes, int n_in,
                              void* d_out, int out_size, void* d_ws,
                              size_t ws_size, hipStream_t stream) {
  const float* q         = (const float*)d_in[0];
  const float* k         = (const float*)d_in[1];
  const float* v         = (const float*)d_in[2];
  const float* attn_bias = (const float*)d_in[3];
  const unsigned char* attn_mask = (const unsigned char*)d_in[4];  // bool
  const float* Wq = (const float*)d_in[5];
  const float* bq = (const float*)d_in[6];
  const float* Wk = (const float*)d_in[7];
  const float* bk = (const float*)d_in[8];
  const float* Wv = (const float*)d_in[9];
  const float* bv = (const float*)d_in[10];
  const float* Wo = (const float*)d_in[11];
  const float* bo = (const float*)d_in[12];

  float* out0   = (float*)d_out;                 // [B,S,512]
  float* scores = out0 + OUT0_ELEMS;             // [B,H,S,S]

  _Float16* Qh  = (_Float16*)d_ws;               // [B,H,S,D]
  _Float16* Kh  = Qh + HALF_MAT;                 // [B,H,S,D]
  _Float16* VT  = Kh + HALF_MAT;                 // [B,H,D,S]
  _Float16* Ctx = VT + HALF_MAT;                 // [4096,512]

  // 1) projections (z: 0=Q, 1=K, 2=V)
  k_qkv_proj<<<dim3(MTOT / 64, HID / 32, 3), 256, 0, stream>>>(
      q, k, v, Wq, bq, Wk, bk, Wv, bv, Qh, Kh, VT);
  // 2) raw masked/biased scores
  k_scores<<<dim3(SEQ / 64, SEQ / 64, BATCH * NHEADS), 256, 0, stream>>>(
      Qh, Kh, attn_bias, attn_mask, scores);
  // 3) softmax in place
  k_softmax<<<dim3(BATCH * NHEADS * SEQ), 256, 0, stream>>>(scores);
  // 4) ctx = P·V
  k_ctx<<<dim3(SEQ / 64, DHEAD / 32, BATCH * NHEADS), 256, 0, stream>>>(
      scores, VT, Ctx);
  // 5) output projection
  k_out_proj<<<dim3(MTOT / 64, HID / 32, 1), 256, 0, stream>>>(
      Ctx, Wo, bo, out0);
}
